// DGCN_6811818131849
// MI455X (gfx1250) — matrix operand order repaired and checked by reading, not compiled, not run on hardware
//
#include <hip/hip_runtime.h>

typedef __attribute__((ext_vector_type(16))) __bf16 v16bf;
typedef __attribute__((ext_vector_type(8)))  __bf16 v8bf;
typedef __attribute__((ext_vector_type(8)))  float  v8f;

#define OUTC 128   // per-direction output channels
#define LDO  256   // concatenated row stride

static __device__ __forceinline__ __bf16 f2bf(float f) {
    union { float f; unsigned u; } v; v.f = f;
    unsigned r = v.u + 0x7FFFu + ((v.u >> 16) & 1u);   // round-nearest-even
    unsigned short h = (unsigned short)(r >> 16);
    __bf16 o; __builtin_memcpy(&o, &h, 2);
    return o;
}

// ---------------- prep kernels ----------------

__global__ void dgcn_zero_deg(unsigned* __restrict__ degS, unsigned* __restrict__ degD, int n) {
    int i = blockIdx.x * blockDim.x + threadIdx.x;
    if (i < n) { degS[i] = 0u; degD[i] = 0u; }
}

__global__ void dgcn_count_deg(const int* __restrict__ src, const int* __restrict__ dst,
                               unsigned* __restrict__ degS, unsigned* __restrict__ degD, int E) {
    int e = blockIdx.x * blockDim.x + threadIdx.x;
    if (e < E) {
        atomicAdd(&degS[src[e]], 1u);
        atomicAdd(&degD[dst[e]], 1u);
    }
}

__global__ void dgcn_make_dinv(const unsigned* __restrict__ deg, float* __restrict__ dinv, int n) {
    int i = blockIdx.x * blockDim.x + threadIdx.x;
    if (i < n) dinv[i] = rsqrtf((float)deg[i] + 1.0f);   // +1 for self-loop
}

// W[K,128] f32 (row-major) -> Wt[128,K] bf16 (transposed)
__global__ void dgcn_transpose_w(const float* __restrict__ W, __bf16* __restrict__ Wt, int K) {
    int i = blockIdx.x * blockDim.x + threadIdx.x;
    if (i < K * OUTC) {
        int k = i >> 7, n = i & 127;
        Wt[(size_t)n * K + k] = f2bf(W[i]);
    }
}

// f32 -> bf16 (optional fused ReLU)
__global__ void dgcn_convert(const float* __restrict__ in, __bf16* __restrict__ out,
                             long n, int relu) {
    long i = (long)blockIdx.x * blockDim.x + threadIdx.x;
    if (i < n) {
        float v = in[i];
        if (relu) v = fmaxf(v, 0.0f);
        out[i] = f2bf(v);
    }
}

// ---------------- WMMA GEMM: C[Npad,128] = A[Npad,K](bf16) * Wt^T ----------------
// block = 256 threads = 8 waves; wave w -> column tile w; block covers 64 rows.
// Npad is a multiple of 64 -> completely branch-free.
__global__ __launch_bounds__(256)
void dgcn_gemm_wmma(const __bf16* __restrict__ A, const __bf16* __restrict__ Wt,
                    float* __restrict__ C, int K) {
    const int lane = threadIdx.x & 31;
    const int wave = threadIdx.x >> 5;          // ntile 0..7
    const int m0   = blockIdx.x * 64;
    const int lm   = lane & 15;
    const int hi   = lane >> 4;                 // 0 or 1

    v8f zero = {};
    v8f acc0 = zero, acc1 = zero, acc2 = zero, acc3 = zero;

    // B fragment base: row (col of W) = wave*16 + lm ; K-half select by lane group
    const __bf16* brow = Wt + (size_t)(wave * 16 + lm) * K + hi * 8;
    const __bf16* arow = A + (size_t)(m0 + lm) * K + hi * 8;
    const size_t tileStride = (size_t)16 * K;   // 16 rows

    for (int k0 = 0; k0 < K; k0 += 32) {
        v16bf bfrag = __builtin_shufflevector(*(const v8bf*)(brow + k0), *(const v8bf*)(brow + k0 + 16),
                                              0,1,2,3,4,5,6,7,8,9,10,11,12,13,14,15);     // k = 8h+s, 16+8h+s

        const __bf16* ap0 = arow + k0;
        const __bf16* ap1 = ap0 + tileStride;
        const __bf16* ap2 = ap1 + tileStride;
        const __bf16* ap3 = ap2 + tileStride;

        v16bf a0 = __builtin_shufflevector(*(const v8bf*)ap0, *(const v8bf*)(ap0 + 16),
                                           0,1,2,3,4,5,6,7,8,9,10,11,12,13,14,15);
        v16bf a1 = __builtin_shufflevector(*(const v8bf*)ap1, *(const v8bf*)(ap1 + 16),
                                           0,1,2,3,4,5,6,7,8,9,10,11,12,13,14,15);
        v16bf a2 = __builtin_shufflevector(*(const v8bf*)ap2, *(const v8bf*)(ap2 + 16),
                                           0,1,2,3,4,5,6,7,8,9,10,11,12,13,14,15);
        v16bf a3 = __builtin_shufflevector(*(const v8bf*)ap3, *(const v8bf*)(ap3 + 16),
                                           0,1,2,3,4,5,6,7,8,9,10,11,12,13,14,15);

        acc0 = __builtin_amdgcn_wmma_f32_16x16x32_bf16(false, a0, false, bfrag, (short)0, acc0, false, false);
        acc1 = __builtin_amdgcn_wmma_f32_16x16x32_bf16(false, a1, false, bfrag, (short)0, acc1, false, false);
        acc2 = __builtin_amdgcn_wmma_f32_16x16x32_bf16(false, a2, false, bfrag, (short)0, acc2, false, false);
        acc3 = __builtin_amdgcn_wmma_f32_16x16x32_bf16(false, a3, false, bfrag, (short)0, acc3, false, false);
    }

    // C layout: element r of acc -> row M = r (lanes 0-15) / 8+r (lanes 16-31), col = lane&15
    const int col = wave * 16 + lm;
    float* c0 = C + (size_t)(m0 + hi * 8) * OUTC + col;
#pragma unroll
    for (int r = 0; r < 8; ++r) c0[(size_t)r * OUTC] = acc0[r];
    float* c1 = c0 + (size_t)16 * OUTC;
#pragma unroll
    for (int r = 0; r < 8; ++r) c1[(size_t)r * OUTC] = acc1[r];
    float* c2 = c1 + (size_t)16 * OUTC;
#pragma unroll
    for (int r = 0; r < 8; ++r) c2[(size_t)r * OUTC] = acc2[r];
    float* c3 = c2 + (size_t)16 * OUTC;
#pragma unroll
    for (int r = 0; r < 8; ++r) c3[(size_t)r * OUTC] = acc3[r];
}

// out[i, coloff+c] = bias[c] + tmp[i,c] * dinv[i]^2   (self-loop term + bias)
__global__ void dgcn_init_out(const float* __restrict__ tmp, const float* __restrict__ bias,
                              const float* __restrict__ dinv, float* __restrict__ out,
                              int N, int coloff) {
    long i = (long)blockIdx.x * blockDim.x + threadIdx.x;
    if (i < (long)N * OUTC) {
        int row = (int)(i >> 7), c = (int)(i & 127);
        float dv = dinv[row];
        out[(size_t)row * LDO + coloff + c] = bias[c] + tmp[i] * dv * dv;
    }
}

// one wave per edge: out[d] += tmp[s] * dinv[s]*dinv[d]; 4 channels per lane
__global__ __launch_bounds__(256)
void dgcn_edge_scatter(const float* __restrict__ tmp, const int* __restrict__ s,
                       const int* __restrict__ d, const float* __restrict__ dinv,
                       float* __restrict__ out, int E, int coloff) {
    int wave = (blockIdx.x << 3) | (threadIdx.x >> 5);
    if (wave >= E) return;
    int lane = threadIdx.x & 31;
    int sn = s[wave], dn = d[wave];
    float norm = dinv[sn] * dinv[dn];
    const float4 v = *(const float4*)(tmp + (size_t)sn * OUTC + lane * 4);
    float* p = out + (size_t)dn * LDO + coloff + lane * 4;
    unsafeAtomicAdd(p + 0, v.x * norm);
    unsafeAtomicAdd(p + 1, v.y * norm);
    unsafeAtomicAdd(p + 2, v.z * norm);
    unsafeAtomicAdd(p + 3, v.w * norm);
}

// ---------------- host ----------------

extern "C" void kernel_launch(void* const* d_in, const int* in_sizes, int n_in,
                              void* d_out, int out_size, void* d_ws, size_t ws_size,
                              hipStream_t stream) {
    const int N = in_sizes[0] / 128;
    const int E = in_sizes[1] / 2;
    const int Npad = (N + 63) & ~63;      // branch-free GEMM over padded rows

    const float* x   = (const float*)d_in[0];
    const int*   src = (const int*)d_in[1];
    const int*   dst = src + E;

    const float* W[6] = { (const float*)d_in[2],  (const float*)d_in[4],
                          (const float*)d_in[6],  (const float*)d_in[8],
                          (const float*)d_in[10], (const float*)d_in[12] };
    const float* B[6] = { (const float*)d_in[3],  (const float*)d_in[5],
                          (const float*)d_in[7],  (const float*)d_in[9],
                          (const float*)d_in[11], (const float*)d_in[13] };
    const int Ks[6] = { 128, 128, 256, 256, 256, 256 };

    // workspace carve-out
    unsigned char* p = (unsigned char*)d_ws;
    auto carve = [&](size_t bytes) -> void* {
        void* r = (void*)p;
        p += (bytes + 255) & ~(size_t)255;
        return r;
    };
    unsigned* degS = (unsigned*)carve((size_t)N * 4);
    unsigned* degD = (unsigned*)carve((size_t)N * 4);
    float* dinvS   = (float*)carve((size_t)N * 4);
    float* dinvD   = (float*)carve((size_t)N * 4);
    __bf16* Wt[6];
    for (int i = 0; i < 6; ++i) Wt[i] = (__bf16*)carve((size_t)Ks[i] * OUTC * 2);
    __bf16* hbf  = (__bf16*)carve((size_t)Npad * LDO * 2);   // padded rows: garbage, never consumed
    float*  hF32 = (float*)carve((size_t)N * LDO * 4);
    float*  tmp  = (float*)carve((size_t)Npad * OUTC * 4);   // padded rows: garbage, never read

    // degrees + inverse-sqrt norms (computed once, reused by all layers)
    dgcn_zero_deg<<<(N + 255) / 256, 256, 0, stream>>>(degS, degD, N);
    dgcn_count_deg<<<(E + 255) / 256, 256, 0, stream>>>(src, dst, degS, degD, E);
    dgcn_make_dinv<<<(N + 255) / 256, 256, 0, stream>>>(degD, dinvD, N);  // forward (dst-deg)
    dgcn_make_dinv<<<(N + 255) / 256, 256, 0, stream>>>(degS, dinvS, N);  // backward (src-deg)

    // weight prep (transpose + bf16)
    for (int i = 0; i < 6; ++i)
        dgcn_transpose_w<<<(Ks[i] * OUTC + 255) / 256, 256, 0, stream>>>(W[i], Wt[i], Ks[i]);

    // input -> bf16
    dgcn_convert<<<((long)N * 128 + 255) / 256, 256, 0, stream>>>(x, hbf, (long)N * 128, 0);

    const int gemmBlocks = Npad / 64;
    const int initBlocks = (int)(((long)N * OUTC + 255) / 256);
    const int edgeBlocks = (E + 7) / 8;

    for (int layer = 0; layer < 3; ++layer) {
        int K = (layer == 0) ? 128 : 256;
        float* out = (layer == 2) ? (float*)d_out : hF32;
        const __bf16* Wf = Wt[2 * layer];
        const __bf16* Wb = Wt[2 * layer + 1];

        // forward conv: src -> dst, dst-based degree
        dgcn_gemm_wmma<<<gemmBlocks, 256, 0, stream>>>(hbf, Wf, tmp, K);
        dgcn_init_out<<<initBlocks, 256, 0, stream>>>(tmp, B[2 * layer], dinvD, out, N, 0);
        dgcn_edge_scatter<<<edgeBlocks, 256, 0, stream>>>(tmp, src, dst, dinvD, out, E, 0);

        // backward conv: dst -> src, src-based degree
        dgcn_gemm_wmma<<<gemmBlocks, 256, 0, stream>>>(hbf, Wb, tmp, K);
        dgcn_init_out<<<initBlocks, 256, 0, stream>>>(tmp, B[2 * layer + 1], dinvS, out, N, 128);
        dgcn_edge_scatter<<<edgeBlocks, 256, 0, stream>>>(tmp, dst, src, dinvS, out, E, 128);

        if (layer != 2)  // ReLU fused into next layer's bf16 conversion
            dgcn_convert<<<((long)N * LDO + 255) / 256, 256, 0, stream>>>(
                out, hbf, (long)N * LDO, 1);
    }
}
